// MaskedMultiHeadAttention_87514253623360
// MI455X (gfx1250) — matrix operand order repaired and checked by reading, not compiled, not run on hardware
//
#include <hip/hip_runtime.h>
#include <hip/hip_bf16.h>
#include <cstdint>

// ---------------------------------------------------------------------------
// Causal MHA for MI455X (gfx1250, wave32, WMMA + TDM + async-LDS copies).
// Compute-bound (~100 GFLOP vs tens of MB @ 23.3 TB/s): all GEMMs (QKV proj,
// QK^T, P*V, out proj) run on v_wmma_f32_16x16x32_f16 with f32 accumulation.
// Contiguous Q/K tiles are DMA'd into LDS by the Tensor Data Mover
// (TENSORcnt); the out-proj A tile uses global_load_async_to_lds (ASYNCcnt).
// ---------------------------------------------------------------------------

typedef __attribute__((ext_vector_type(16))) _Float16 v16h;
typedef __attribute__((ext_vector_type(8)))  _Float16 v8h;
typedef __attribute__((ext_vector_type(8)))  float    v8f;
typedef __attribute__((ext_vector_type(4)))  float    v4f;
typedef unsigned int u32x4 __attribute__((ext_vector_type(4)));
typedef int          i32x4 __attribute__((ext_vector_type(4)));
typedef int          i32x8 __attribute__((ext_vector_type(8)));

namespace {
constexpr int S_LEN = 4096;
constexpr int NH    = 16;
constexpr int DH    = 64;
constexpr int DM    = 1024;
constexpr int DQKV  = 3072;
constexpr float ATTN_SCALE = 0.125f;  // 1/sqrt(64)
}

// ---- WMMA fragment helpers (wave32 layouts per CDNA5 ISA 7.12.2) ----------

// A fragment: 16x32 f16, row-major tile in LDS, leading dim `ld` (halves,
// multiple of 8). lanes 0-15 hold row M=lane, K in {0..7, 16..23};
// lanes 16-31 hold row M=lane-16, K in {8..15, 24..31}.
__device__ __forceinline__ v16h frag_a_f16(const _Float16* p, int ld, int coloff) {
  const int lane = threadIdx.x & 31;
  const int m    = lane & 15;
  const int half = lane >> 4;
  const _Float16* row = p + m * ld + coloff;
  v8h lo = *reinterpret_cast<const v8h*>(row + 8 * half);
  v8h hi = *reinterpret_cast<const v8h*>(row + 16 + 8 * half);
  v16h r;
#pragma unroll
  for (int i = 0; i < 8; ++i) { r[i] = lo[i]; r[8 + i] = hi[i]; }
  return r;
}

// B fragment: 32x16 f16 with B stored TRANSPOSED in LDS as Bt[N][K],
// leading dim `ld` (halves, multiple of 8). lane n = lane&15 reads
// K in {8h..8h+7, 16+8h..16+8h+7} (h = lane>>4), matching the A fragment.
__device__ __forceinline__ v16h frag_b_f16(const _Float16* bt, int ld, int coloff) {
  const int lane = threadIdx.x & 31;
  const int n    = lane & 15;
  const int kb   = (lane >> 4) * 8;
  const _Float16* src = bt + n * ld + coloff + kb;
  v8h lo = *reinterpret_cast<const v8h*>(src);
  v8h hi = *reinterpret_cast<const v8h*>(src + 16);
  v16h r;
#pragma unroll
  for (int i = 0; i < 8; ++i) { r[i] = lo[i]; r[8 + i] = hi[i]; }
  return r;
}

__device__ __forceinline__ v8f wmma16x16x32(v16h a, v16h b, v8f c) {
  return __builtin_amdgcn_wmma_f32_16x16x32_f16(false, a, false, b,
                                                (short)0, c, false, false);
}

// ---- Tensor Data Mover: 1-D contiguous copy global -> LDS -----------------
// D# per CDNA5 ISA: count=1, type=2, data_size=1 (2 bytes),
// tile = 1 row of n f16 elements (n <= 65535, n*2 multiple of 4 bytes).
__device__ __forceinline__ void tdm_copy_1d(unsigned ldsOff,
                                            unsigned long long gaddr,
                                            unsigned n) {
  u32x4 g0;
  g0[0] = 1u;                                            // count=1 (valid)
  g0[1] = ldsOff;                                        // lds_addr (bytes)
  g0[2] = (unsigned)(gaddr & 0xFFFFFFFFull);             // global_addr lo
  g0[3] = (unsigned)((gaddr >> 32) & 0x01FFFFFFull)      // global_addr hi
          | 0x80000000u;                                 // type=2 ("image")
  i32x8 g1;
  g1[0] = 0x00010000;                                    // data_size=1 (2B)
  g1[1] = (int)((n & 0xFFFFu) << 16);                    // tensor_dim0[15:0]
  g1[2] = (int)(((n >> 16) & 0xFFFFu) | (1u << 16));     // dim0 hi | dim1=1
  g1[3] = (int)((n & 0xFFFFu) << 16);                    // tile_dim0=n
  g1[4] = 1;                                             // tile_dim1=1
  g1[5] = (int)n;                                        // dim0_stride lo
  g1[6] = 0;
  g1[7] = 0;
  i32x4 z4 = {0, 0, 0, 0};
#if __clang_major__ >= 23
  i32x8 z8 = {0, 0, 0, 0, 0, 0, 0, 0};
  __builtin_amdgcn_tensor_load_to_lds(g0, g1, z4, z4, z8, 0);
#else
  __builtin_amdgcn_tensor_load_to_lds(g0, g1, z4, z4, 0);
#endif
}

__device__ __forceinline__ unsigned lds_offset(const void* p) {
  return (unsigned)(uintptr_t)p;  // generic LDS ptr: low 32 bits = LDS offset
}

// Per-lane async 16-byte copy global -> LDS (tracked with ASYNCcnt).
__device__ __forceinline__ void async_copy_b128(unsigned ldsOff,
                                                unsigned long long gaddr) {
  asm volatile("global_load_async_to_lds_b128 %0, %1, off"
               :: "v"(ldsOff), "v"(gaddr) : "memory");
}

// ---------------------------------------------------------------------------
// Kernel 1: QKV projection. C[4096,3072] = x[4096,1024] @ w[1024,3072] + b,
// scattered into head-major f16 Q/K/V [H][S][64].
// Block tile 128x64, BK=32, 8 waves -> each wave 32x32 (2x2 WMMA tiles).
// (f32 -> f16 conversion forces the VGPR bounce here; no async path.)
// ---------------------------------------------------------------------------
__global__ __launch_bounds__(256) void qkv_proj_kernel(
    const float* __restrict__ x, const float* __restrict__ w,
    const float* __restrict__ bias,
    _Float16* __restrict__ Qh, _Float16* __restrict__ Kh,
    _Float16* __restrict__ Vh) {
  __shared__ alignas(16) _Float16 As[128 * 32];   // A tile, row-major [m][k]
  __shared__ alignas(16) _Float16 Bts[64 * 32];   // B tile transposed [n][k]

  const int tid  = threadIdx.x;
  const int wave = tid >> 5;
  const int lane = tid & 31;
  const int n0   = blockIdx.x * 64;    // over 3072
  const int m0   = blockIdx.y * 128;   // over 4096
  const int wm   = wave >> 1;          // 0..3 -> rows wm*32
  const int wn   = wave & 1;           // 0..1 -> cols wn*32

  v8f acc[2][2] = {};

  for (int kt = 0; kt < DM; kt += 32) {
    // A tile: 128x32 f32 -> f16. 16 elems/thread (half a row).
    {
      const int idx = tid * 16;
      const int r = idx >> 5, c = idx & 31;
      const v4f* src = reinterpret_cast<const v4f*>(
          x + (size_t)(m0 + r) * DM + kt + c);
      _Float16* dst = As + r * 32 + c;
#pragma unroll
      for (int i = 0; i < 4; ++i) {
        v4f v = src[i];
#pragma unroll
        for (int j = 0; j < 4; ++j) dst[i * 4 + j] = (_Float16)v[j];
      }
    }
    // B tile: 32(k) x 64(n) f32 -> f16, stored transposed. 8 elems/thread.
    {
      const int idx = tid * 8;
      const int kk = idx >> 6, nn = idx & 63;
      const v4f* src = reinterpret_cast<const v4f*>(
          w + (size_t)(kt + kk) * DQKV + n0 + nn);
#pragma unroll
      for (int i = 0; i < 2; ++i) {
        v4f v = src[i];
#pragma unroll
        for (int j = 0; j < 4; ++j)
          Bts[(nn + i * 4 + j) * 32 + kk] = (_Float16)v[j];
      }
    }
    __syncthreads();

    v16h aF[2], bF[2];
#pragma unroll
    for (int t = 0; t < 2; ++t) {
      aF[t] = frag_a_f16(As + (wm * 32 + t * 16) * 32, 32, 0);
      bF[t] = frag_b_f16(Bts + (wn * 32 + t * 16) * 32, 32, 0);
    }
#pragma unroll
    for (int tm = 0; tm < 2; ++tm)
#pragma unroll
      for (int tn = 0; tn < 2; ++tn)
        acc[tm][tn] = wmma16x16x32(aF[tm], bF[tn], acc[tm][tn]);
    __syncthreads();
  }

  // Epilogue: C-layout (row = 8*(lane>>4)+r, col = lane&15); scatter to QKV.
  const int mbase = (lane >> 4) * 8;
  const int nn    = lane & 15;
#pragma unroll
  for (int tm = 0; tm < 2; ++tm)
#pragma unroll
    for (int tn = 0; tn < 2; ++tn) {
      const int cg   = n0 + wn * 32 + tn * 16 + nn;     // 0..3071
      const float bv = bias[cg];
      const int part = cg >> 10;                        // 0=Q 1=K 2=V
      const int rem  = cg & 1023;
      const int hh   = rem >> 6;
      const int dd   = rem & 63;
      _Float16* dst = (part == 0) ? Qh : (part == 1) ? Kh : Vh;
#pragma unroll
      for (int r = 0; r < 8; ++r) {
        const int rg = m0 + wm * 32 + tm * 16 + mbase + r;
        dst[((size_t)hh * S_LEN + rg) * DH + dd] =
            (_Float16)(acc[tm][tn][r] + bv);
      }
    }
}

// ---------------------------------------------------------------------------
// Kernel 2: causal flash attention, one block per (128-query tile, head).
// 8 waves, each owns 16 query rows. Q tile (16KB) and each 64-key K tile
// (8KB) are contiguous in head-major layout -> TDM DMA into LDS (wave 0,
// TENSORcnt). V tile needs a transpose -> global->VGPR->LDS scatter.
// ---------------------------------------------------------------------------
__global__ __launch_bounds__(256) void attn_kernel(
    const _Float16* __restrict__ Qh, const _Float16* __restrict__ Kh,
    const _Float16* __restrict__ Vh, _Float16* __restrict__ O) {
  __shared__ alignas(16) _Float16 Qs[128 * 64];        // [qrow][dim]
  __shared__ alignas(16) _Float16 Ks[64 * 64];         // [key][dim] == Bt for QK^T
  __shared__ alignas(16) _Float16 Vts[64 * 64];        // [dim][key] == Bt for P*V
  __shared__ alignas(16) _Float16 Ps[8 * 16 * 64];     // per-wave P scratch

  const int tid  = threadIdx.x;
  const int wave = tid >> 5;
  const int lane = tid & 31;
  const int h    = blockIdx.y;
  const int q0   = blockIdx.x * 128;
  const _Float16* Qg = Qh + (size_t)h * S_LEN * DH;
  const _Float16* Kg = Kh + (size_t)h * S_LEN * DH;
  const _Float16* Vg = Vh + (size_t)h * S_LEN * DH;

  // Q tile: one TDM descriptor moves the whole contiguous 128x64 tile.
  if (wave == 0) {
    tdm_copy_1d(lds_offset(Qs),
                (unsigned long long)(uintptr_t)(Qg + (size_t)q0 * DH),
                128 * DH);
    __builtin_amdgcn_s_wait_tensorcnt(0);
  }

  const int mbase = (lane >> 4) * 8;
  const int nn    = lane & 15;
  float mrow[8], lrow[8];
  v8f accO[4] = {};
#pragma unroll
  for (int r = 0; r < 8; ++r) { mrow[r] = -1e30f; lrow[r] = 0.0f; }
  __syncthreads();

  const int nkb = (blockIdx.x + 1) * 2;   // causal: key blocks of 64
  for (int kb = 0; kb < nkb; ++kb) {
    const int k0 = kb * 64;
    // K tile: contiguous 64x64 -> TDM DMA (wave 0 issues + drains TENSORcnt).
    if (wave == 0) {
      tdm_copy_1d(lds_offset(Ks),
                  (unsigned long long)(uintptr_t)(Kg + (size_t)k0 * DH),
                  64 * DH);
      __builtin_amdgcn_s_wait_tensorcnt(0);
    }
    // V tile transposed into LDS: 16 halves/thread, scatter stores.
    {
      const int idx = tid * 16;
      const int r = idx >> 6, c = idx & 63;
      const v8h* vsrc = reinterpret_cast<const v8h*>(Vg + (size_t)(k0 + r) * DH + c);
      v8h v0 = vsrc[0], v1 = vsrc[1];
#pragma unroll
      for (int i = 0; i < 8; ++i) {
        Vts[(c + i) * 64 + r]     = v0[i];
        Vts[(c + 8 + i) * 64 + r] = v1[i];
      }
    }
    __syncthreads();

    // S = Q * K^T (4 key-subtiles of 16, contraction over d=64 in 2 steps).
    v8f sc[4];
#pragma unroll
    for (int t = 0; t < 4; ++t) {
      v8f a = {};
#pragma unroll
      for (int ks = 0; ks < 2; ++ks) {
        v16h af = frag_a_f16(Qs + (wave * 16) * 64, 64, ks * 32);
        v16h bf = frag_b_f16(Ks + (t * 16) * 64, 64, ks * 32);
        a = wmma16x16x32(af, bf, a);
      }
      sc[t] = a;
    }

    // Scale + causal mask + row-max (rows live at mbase+r, col = nn).
    float rmax[8];
#pragma unroll
    for (int r = 0; r < 8; ++r) rmax[r] = -1e30f;
#pragma unroll
    for (int t = 0; t < 4; ++t) {
      const int kg = k0 + t * 16 + nn;
#pragma unroll
      for (int r = 0; r < 8; ++r) {
        const int qg = q0 + wave * 16 + mbase + r;
        float s = sc[t][r] * ATTN_SCALE;
        if (kg > qg) s = -1e30f;
        sc[t][r] = s;
        rmax[r] = fmaxf(rmax[r], s);
      }
    }
    for (int msk = 1; msk < 16; msk <<= 1)
#pragma unroll
      for (int r = 0; r < 8; ++r)
        rmax[r] = fmaxf(rmax[r], __shfl_xor(rmax[r], msk, 32));

    // Online softmax update.
    float alpha[8], rsum[8];
#pragma unroll
    for (int r = 0; r < 8; ++r) {
      const float mn = fmaxf(mrow[r], rmax[r]);
      alpha[r] = __expf(mrow[r] - mn);
      mrow[r]  = mn;
      rsum[r]  = 0.0f;
    }
#pragma unroll
    for (int t = 0; t < 4; ++t)
#pragma unroll
      for (int r = 0; r < 8; ++r) {
        const float p = __expf(sc[t][r] - mrow[r]);
        rsum[r] += p;
        Ps[wave * 1024 + (mbase + r) * 64 + t * 16 + nn] = (_Float16)p;
      }
    for (int msk = 1; msk < 16; msk <<= 1)
#pragma unroll
      for (int r = 0; r < 8; ++r) rsum[r] += __shfl_xor(rsum[r], msk, 32);
#pragma unroll
    for (int r = 0; r < 8; ++r) lrow[r] = lrow[r] * alpha[r] + rsum[r];
#pragma unroll
    for (int t = 0; t < 4; ++t)
#pragma unroll
      for (int r = 0; r < 8; ++r) accO[t][r] *= alpha[r];

    // Same-wave LDS RAW on Ps: drain the CDNA5 DS counter.
    asm volatile("s_wait_dscnt 0" ::: "memory");

    // O += P * V (4 dim-subtiles, contraction over 64 keys in 2 steps).
#pragma unroll
    for (int t = 0; t < 4; ++t)
#pragma unroll
      for (int ks = 0; ks < 2; ++ks) {
        v16h pf = frag_a_f16(Ps + wave * 1024, 64, ks * 32);
        v16h vf = frag_b_f16(Vts + (t * 16) * 64, 64, ks * 32);
        accO[t] = wmma16x16x32(pf, vf, accO[t]);
      }
    __syncthreads();
  }

  // Normalize and emit O as f16 [S][DM] so the out-proj consumes it directly.
#pragma unroll
  for (int t = 0; t < 4; ++t)
#pragma unroll
    for (int r = 0; r < 8; ++r) {
      const int qg = q0 + wave * 16 + mbase + r;
      const float o = accO[t][r] / lrow[r];
      O[(size_t)qg * DM + h * DH + t * 16 + nn] = (_Float16)o;
    }
}

// ---------------------------------------------------------------------------
// Kernel 3: output projection. out[4096,1024] = O_f16 @ w_out + b_out (f32).
// A tile is already f16 -> async global->LDS copies (ASYNCcnt), no VGPR
// bounce. B tile still converts f32->f16 through VGPRs.
// ---------------------------------------------------------------------------
__global__ __launch_bounds__(256) void out_proj_kernel(
    const _Float16* __restrict__ A, const float* __restrict__ w,
    const float* __restrict__ bias, float* __restrict__ out) {
  __shared__ alignas(16) _Float16 As[128 * 32];
  __shared__ alignas(16) _Float16 Bts[64 * 32];

  const int tid  = threadIdx.x;
  const int wave = tid >> 5;
  const int lane = tid & 31;
  const int n0   = blockIdx.x * 64;    // over 1024
  const int m0   = blockIdx.y * 128;   // over 4096
  const int wm   = wave >> 1;
  const int wn   = wave & 1;

  v8f acc[2][2] = {};

  for (int kt = 0; kt < DM; kt += 32) {
    // A tile: 128x32 f16 via async LDS copies (2 x b128 chunks per thread).
    {
#pragma unroll
      for (int cc = 0; cc < 2; ++cc) {
        const int c   = tid + cc * 256;        // chunk id, 512 x 16B = 8KB
        const int r   = c >> 2;                // row 0..127
        const int q   = c & 3;                 // 16B quarter of a 64B row
        async_copy_b128(
            lds_offset(As + r * 32 + q * 8),
            (unsigned long long)(uintptr_t)(A + (size_t)(m0 + r) * DM + kt + q * 8));
      }
    }
    // B tile: 32(k) x 64(n) f32 -> f16, stored transposed. 8 elems/thread.
    {
      const int idx = tid * 8;
      const int kk = idx >> 6, nn = idx & 63;
      const v4f* src = reinterpret_cast<const v4f*>(
          w + (size_t)(kt + kk) * DM + n0 + nn);
#pragma unroll
      for (int i = 0; i < 2; ++i) {
        v4f v = src[i];
#pragma unroll
        for (int j = 0; j < 4; ++j)
          Bts[(nn + i * 4 + j) * 32 + kk] = (_Float16)v[j];
      }
    }
    asm volatile("s_wait_asynccnt 0" ::: "memory");
    __syncthreads();

    v16h aF[2], bF[2];
#pragma unroll
    for (int t = 0; t < 2; ++t) {
      aF[t] = frag_a_f16(As + (wm * 32 + t * 16) * 32, 32, 0);
      bF[t] = frag_b_f16(Bts + (wn * 32 + t * 16) * 32, 32, 0);
    }
#pragma unroll
    for (int tm = 0; tm < 2; ++tm)
#pragma unroll
      for (int tn = 0; tn < 2; ++tn)
        acc[tm][tn] = wmma16x16x32(aF[tm], bF[tn], acc[tm][tn]);
    __syncthreads();
  }

  const int mbase = (lane >> 4) * 8;
  const int nn    = lane & 15;
#pragma unroll
  for (int tm = 0; tm < 2; ++tm)
#pragma unroll
    for (int tn = 0; tn < 2; ++tn) {
      const int cg   = n0 + wn * 32 + tn * 16 + nn;
      const float bv = bias[cg];
#pragma unroll
      for (int r = 0; r < 8; ++r) {
        const int rg = m0 + wm * 32 + tm * 16 + mbase + r;
        out[(size_t)rg * DM + cg] = acc[tm][tn][r] + bv;
      }
    }
}

// ---------------------------------------------------------------------------
extern "C" void kernel_launch(void* const* d_in, const int* in_sizes, int n_in,
                              void* d_out, int out_size, void* d_ws,
                              size_t ws_size, hipStream_t stream) {
  const float* x     = (const float*)d_in[0];
  const float* w_qkv = (const float*)d_in[1];
  const float* b_qkv = (const float*)d_in[2];
  const float* w_out = (const float*)d_in[3];
  const float* b_out = (const float*)d_in[4];
  float* out = (float*)d_out;

  // Workspace: Q,K,V head-major f16 [16][4096][64] + O f16 [4096][1024] = 32MB.
  const size_t headElems = (size_t)NH * S_LEN * DH;  // 4M halves
  _Float16* Qh = (_Float16*)d_ws;
  _Float16* Kh = Qh + headElems;
  _Float16* Vh = Kh + headElems;
  _Float16* Ob = Vh + headElems;

  dim3 blk(256);
  qkv_proj_kernel<<<dim3(DQKV / 64, S_LEN / 128), blk, 0, stream>>>(
      x, w_qkv, b_qkv, Qh, Kh, Vh);
  attn_kernel<<<dim3(S_LEN / 128, NH), blk, 0, stream>>>(Qh, Kh, Vh, Ob);
  out_proj_kernel<<<dim3(DM / 64, S_LEN / 128), blk, 0, stream>>>(
      Ob, w_out, b_out, out);
}
